// StudentInputModule_31439160607015
// MI455X (gfx1250) — matrix operand order repaired and checked by reading, not compiled, not run on hardware
//
#include <hip/hip_runtime.h>

// ---------------------------------------------------------------------------
// Patch-attention pooler for MI455X (gfx1250, wave32, WMMA).
// K/V projections (137 GFLOP) + out projection (4.3 GFLOP) on
// v_wmma_f32_16x16x32_bf16. 4 patches/WG (M-tile 64) x 2 N-tiles per pass:
// each B fragment feeds 4 WMMAs, each A fragment feeds 2 -> 1.5 mem ops/WMMA.
// KV tiles stored column-major in LDS so the epilogue is 4 cvt_pk + 1 b128.
// ---------------------------------------------------------------------------

typedef __bf16 bf16_t;
typedef __attribute__((ext_vector_type(16))) __bf16 v16bf;
typedef __attribute__((ext_vector_type(8)))  __bf16 v8bf;
typedef __attribute__((ext_vector_type(4)))  __bf16 v4bf;
typedef __attribute__((ext_vector_type(8)))  float  v8f;
typedef __attribute__((ext_vector_type(4)))  float  v4f;

#define DMODEL 1024
#define LPATCH 16
#define NHEADS 8
#define DHEAD  128
#define NPATCH 2048            // B*S/L = 8*4096/16
#define KSTEPS (DMODEL / 32)   // 32 wmma K-steps per 1024-deep GEMM
#define NTILES (DMODEL / 16)   // 64 N-tiles per 1024-wide output
#define PPW    4               // patches per workgroup (M-tile = 64 rows)

static constexpr int LDA    = DMODEL + 8;      // A rows: bf16, 16B-aligned, bank spread
static constexpr int PSTRD  = LPATCH * LDA;    // per-patch A tile stride
static constexpr int TSTRD  = LPATCH * DMODEL; // per-patch KVT tile stride (col-major)

// -------------------------- kernel 1: weight cvt ---------------------------
__global__ void cvt_weights_kernel(const float* __restrict__ in_proj_w,
                                   const float* __restrict__ out_w,
                                   bf16_t* __restrict__ wkb,
                                   bf16_t* __restrict__ wvb,
                                   bf16_t* __restrict__ owb) {
    int idx = blockIdx.x * blockDim.x + threadIdx.x;
    if (idx < DMODEL * DMODEL) {
        wkb[idx] = (bf16_t)in_proj_w[1 * DMODEL * DMODEL + idx];  // wk
        wvb[idx] = (bf16_t)in_proj_w[2 * DMODEL * DMODEL + idx];  // wv
        owb[idx] = (bf16_t)out_w[idx];
    }
}

// -------------------------- kernel 2: q projection -------------------------
__global__ void q_proj_kernel(const float* __restrict__ in_proj_w,
                              const float* __restrict__ in_proj_b,
                              const float* __restrict__ patch_query,
                              float* __restrict__ qv) {
    int n = blockIdx.x * blockDim.x + threadIdx.x;
    if (n < DMODEL) {
        const float* row = in_proj_w + (size_t)n * DMODEL;  // wq row n
        float s = in_proj_b[n];
        for (int d = 0; d < DMODEL; ++d) s += patch_query[d] * row[d];
        qv[n] = s;
    }
}

// ---- A-fragment loader: 16x32 A fragment in this code's contiguous k order (B matches).
// lane row = rlo; lanes 0-15 hold K 0..15, lanes 16-31 hold K 16..31.
__device__ __forceinline__ v16bf load_a_frag(const bf16_t* __restrict__ Abuf,
                                             int rlo, int hiA, int kk) {
    const int ka = kk * 32 + hiA * 16;
    const v8bf alo = *(const v8bf*)(&Abuf[rlo * LDA + ka]);
    const v8bf ahi = *(const v8bf*)(&Abuf[rlo * LDA + ka + 8]);
    v16bf a;
#pragma unroll
    for (int i = 0; i < 8; ++i) { a[i] = alo[i]; a[i + 8] = ahi[i]; }
    return a;
}

// ---- one K- or V-projection pass: 4 patch tiles x 64 N-tiles = A @ W^T ----
// Output goes to LDS in tile-column layout: KVT[p*TSTRD + col*16 + m].
__device__ __forceinline__ void kv_gemm_phase(const bf16_t* __restrict__ Asb,
                                              bf16_t* __restrict__ KVT,
                                              const bf16_t* __restrict__ W,
                                              int wave, int rlo, int hiA) {
    for (int j = 0; j < NTILES / 8; j += 2) {       // 4 passes, 2 N-tiles each
        const int ncol0 = (wave + 8 * j) * 16 + rlo;
        const int ncol1 = ncol0 + 128;              // next wave-strided tile
        // B 32x16, contiguous k like A: lanes 0-15 -> N=lane, K 0..15; lanes 16-31 -> N, K 16..31.
        const bf16_t* B0 = W + (size_t)ncol0 * DMODEL + (hiA ? 16 : 0);
        const bf16_t* B1 = W + (size_t)ncol1 * DMODEL + (hiA ? 16 : 0);

        v8f c0[PPW] = {}, c1[PPW] = {};
        for (int kk = 0; kk < KSTEPS; ++kk) {
            const v16bf b0 = *(const v16bf*)(B0 + kk * 32);
            const v16bf b1 = *(const v16bf*)(B1 + kk * 32);
#pragma unroll
            for (int p = 0; p < PPW; ++p) {
                const v16bf a = load_a_frag(Asb + p * PSTRD, rlo, hiA, kk);
                c0[p] = __builtin_amdgcn_wmma_f32_16x16x32_bf16(
                            false, a, false, b0, (short)0, c0[p], false, false);
                c1[p] = __builtin_amdgcn_wmma_f32_16x16x32_bf16(
                            false, a, false, b1, (short)0, c1[p], false, false);
            }
        }
        // D: lanes 0-15 hold M=0..7, lanes 16-31 hold M=8..15 of column ncol.
        // Column-major KVT -> each lane packs 8 bf16 and does ONE b128 store.
#pragma unroll
        for (int p = 0; p < PPW; ++p) {
            v8bf d0, d1;
#pragma unroll
            for (int r = 0; r < 8; ++r) { d0[r] = (bf16_t)c0[p][r]; d1[r] = (bf16_t)c1[p][r]; }
            *(v8bf*)(&KVT[p * TSTRD + ncol0 * LPATCH + hiA * 8]) = d0;
            *(v8bf*)(&KVT[p * TSTRD + ncol1 * LPATCH + hiA * 8]) = d1;
        }
    }
}

// --------------- kernel 3: fused gather + K/V WMMA + attention -------------
__global__ __launch_bounds__(256, 1)
void patch_attn_kernel(const int* __restrict__ tokens,
                       const unsigned char* __restrict__ amask,
                       const float* __restrict__ emb,
                       const float* __restrict__ qv,
                       const bf16_t* __restrict__ wkb,
                       const bf16_t* __restrict__ wvb,
                       bf16_t* __restrict__ ctxb,
                       float* __restrict__ pvf,
                       float* __restrict__ pv_out) {
    __shared__ bf16_t Asb[PPW * PSTRD];     // gathered tokens (bf16, row-major) ~129 KB
    __shared__ bf16_t KVT[PPW * TSTRD];     // K then V tiles (column-major)      128 KB
    __shared__ float  qsh[DMODEL];          // projected query                      4 KB
    __shared__ float  attw[PPW * NHEADS * LPATCH];  // softmax weights              2 KB
    __shared__ int    toks[PPW * LPATCH];
    __shared__ unsigned int maskbits[PPW];

    const int pbase = blockIdx.x * PPW;
    const int tid   = threadIdx.x;
    const int lane  = tid & 31;
    const int wave  = tid >> 5;
    const int rlo   = lane & 15;
    const int hiA   = lane >> 4;            // 0: lanes 0-15, 1: lanes 16-31

    if (tid < PPW * LPATCH) toks[tid] = tokens[pbase * LPATCH + tid];
    if (tid < PPW) {
        const int patch = pbase + tid;
        unsigned int m = 0;
        for (int l = 0; l < LPATCH; ++l)
            if (amask[patch * LPATCH + l]) m |= (1u << l);
        const bool valid = (m != 0);
        if (!valid) m = 1u;                 // safe-softmax: force position 0
        maskbits[tid] = m;
        const float pv = valid ? 1.0f : 0.0f;
        pvf[patch]    = pv;
        pv_out[patch] = pv;
    }
    __syncthreads();

    // ---- stage A: gather 64 embedding rows, float4 -> 4x bf16 per op ----
    for (int i = tid; i < PPW * LPATCH * DMODEL / 4; i += 256) {
        const int e   = i * 4;
        const int row = e >> 10;            // 0..63 (patch*16 + l)
        const int col = e & (DMODEL - 1);
        const v4f x = *(const v4f*)(&emb[(size_t)toks[row] * DMODEL + col]);
        v4bf y;
#pragma unroll
        for (int t = 0; t < 4; ++t) y[t] = (bf16_t)x[t];
        *(v4bf*)(&Asb[(row >> 4) * PSTRD + (row & 15) * LDA + col]) = y;
    }
    for (int i = tid; i < DMODEL; i += 256) qsh[i] = qv[i];
    __syncthreads();

    // ================= phase 1: K = A @ wk^T (M-tile 64) =================
    kv_gemm_phase(Asb, KVT, wkb, wave, rlo, hiA);
    __syncthreads();

    // ============ phase 2: scores + masked softmax (wave == head) ========
    const int h = wave;
    for (int p = 0; p < PPW; ++p) {
        const unsigned int safe = maskbits[p];
        const bool on = (lane < LPATCH) && (safe & (1u << lane));
        float score = -__builtin_inff();
        if (on) {
            float s = 0.f;
            const bf16_t* kp = &KVT[p * TSTRD + lane];     // K[l=lane][d] at d*16
            for (int d = 0; d < DHEAD; ++d)
                s += qsh[h * DHEAD + d] * (float)kp[(h * DHEAD + d) * LPATCH];
            score = s * 0.08838834764831845f;   // 1/sqrt(128)
        }
        float mx = score;
        for (int off = 8; off >= 1; off >>= 1) mx = fmaxf(mx, __shfl_xor(mx, off, 16));
        const float e = on ? __expf(score - mx) : 0.f;
        float ssum = e;
        for (int off = 8; off >= 1; off >>= 1) ssum += __shfl_xor(ssum, off, 16);
        if (lane < LPATCH)
            attw[(p * NHEADS + h) * LPATCH + lane] = e / ssum;
    }
    __syncthreads();

    // ================= phase 3: V = A @ wv^T (reuse KVT buffer) ==========
    kv_gemm_phase(Asb, KVT, wvb, wave, rlo, hiA);
    __syncthreads();

    // ================= phase 4: ctx = attn @ V, store bf16 ===============
    for (int p = 0; p < PPW; ++p) {
        float acc0 = 0.f, acc1 = 0.f, acc2 = 0.f, acc3 = 0.f;
        const float* wrow = &attw[(p * NHEADS + h) * LPATCH];
        const bf16_t* vt = &KVT[p * TSTRD];
        for (int l = 0; l < LPATCH; ++l) {
            const float wl = wrow[l];                      // LDS broadcast
            // V[l][hd] at KVT[hd*16 + l]
            acc0 += wl * (float)vt[(h * DHEAD +  0 + lane) * LPATCH + l];
            acc1 += wl * (float)vt[(h * DHEAD + 32 + lane) * LPATCH + l];
            acc2 += wl * (float)vt[(h * DHEAD + 64 + lane) * LPATCH + l];
            acc3 += wl * (float)vt[(h * DHEAD + 96 + lane) * LPATCH + l];
        }
        bf16_t* crow = ctxb + (size_t)(pbase + p) * DMODEL + h * DHEAD;
        crow[lane]      = (bf16_t)acc0;
        crow[32 + lane] = (bf16_t)acc1;
        crow[64 + lane] = (bf16_t)acc2;
        crow[96 + lane] = (bf16_t)acc3;
    }
}

// ------------- kernel 4: out projection GEMM [2048,1024]x[1024,1024] -------
__global__ __launch_bounds__(256, 1)
void out_proj_kernel(const bf16_t* __restrict__ ctxb,
                     const bf16_t* __restrict__ owb,
                     const float* __restrict__ out_b,
                     const float* __restrict__ pvf,
                     float* __restrict__ out) {
    const int tid  = threadIdx.x;
    const int lane = tid & 31;
    const int wave = tid >> 5;
    const int rlo  = lane & 15;
    const int hiA  = lane >> 4;
    const int mtile = blockIdx.x;                   // 0..127 (2048 rows / 16)
    const int rowA  = mtile * 16 + rlo;

    const bf16_t* Abase = ctxb + (size_t)rowA * DMODEL + (hiA ? 16 : 0);

    for (int j = 0; j < 8; ++j) {
        const int ntile = wave + 8 * j;             // 0..63
        const int ncol  = ntile * 16 + rlo;
        const bf16_t* Bbase = owb + (size_t)ncol * DMODEL + (hiA ? 16 : 0);

        v8f c = {};
        for (int kk = 0; kk < KSTEPS; ++kk) {
            const v8bf alo = *(const v8bf*)(Abase + kk * 32);
            const v8bf ahi = *(const v8bf*)(Abase + kk * 32 + 8);
            v16bf a;
#pragma unroll
            for (int i = 0; i < 8; ++i) { a[i] = alo[i]; a[i + 8] = ahi[i]; }
            const v16bf b = *(const v16bf*)(Bbase + kk * 32);
            c = __builtin_amdgcn_wmma_f32_16x16x32_bf16(
                    false, a, false, b, (short)0, c, false, false);
        }
        const float bias = out_b[ncol];
        for (int r = 0; r < 8; ++r) {
            const int m = mtile * 16 + r + (hiA ? 8 : 0);
            out[(size_t)m * DMODEL + ncol] = (c[r] + bias) * pvf[m];
        }
    }
}

// ------------------------------- launcher ----------------------------------
extern "C" void kernel_launch(void* const* d_in, const int* in_sizes, int n_in,
                              void* d_out, int out_size, void* d_ws, size_t ws_size,
                              hipStream_t stream) {
    // setup_inputs() order
    const int*           tokens      = (const int*)d_in[0];
    const unsigned char* amask       = (const unsigned char*)d_in[1];
    const float*         emb         = (const float*)d_in[2];
    const float*         patch_query = (const float*)d_in[3];
    const float*         in_proj_w   = (const float*)d_in[4];
    const float*         in_proj_b   = (const float*)d_in[5];
    const float*         out_w       = (const float*)d_in[6];
    const float*         out_b       = (const float*)d_in[7];

    // workspace layout (bytes)
    char* ws = (char*)d_ws;
    const size_t WMAT = (size_t)DMODEL * DMODEL;    // 1M elements
    bf16_t* wkb  = (bf16_t*)(ws);                                   // 2 MB
    bf16_t* wvb  = (bf16_t*)(ws + 2 * WMAT * sizeof(bf16_t));       // 2 MB
    bf16_t* owb  = (bf16_t*)(ws + 4 * WMAT * sizeof(bf16_t));       // 2 MB
    float*  qv   = (float*)(ws + 6 * WMAT * sizeof(bf16_t));        // 4 KB
    float*  pvf  = (float*)((char*)qv + DMODEL * sizeof(float));    // 8 KB
    bf16_t* ctxb = (bf16_t*)((char*)pvf + NPATCH * sizeof(float));  // 4 MB

    float* latent_out = (float*)d_out;                        // [2048,1024] f32
    float* pv_out     = latent_out + (size_t)NPATCH * DMODEL; // [2048] 0/1 floats

    cvt_weights_kernel<<<(DMODEL * DMODEL + 255) / 256, 256, 0, stream>>>(
        in_proj_w, out_w, wkb, wvb, owb);
    q_proj_kernel<<<(DMODEL + 255) / 256, 256, 0, stream>>>(
        in_proj_w, in_proj_b, patch_query, qv);
    patch_attn_kernel<<<NPATCH / PPW, 256, 0, stream>>>(
        tokens, amask, emb, qv, wkb, wvb, ctxb, pvf, pv_out);
    out_proj_kernel<<<NPATCH / 16, 256, 0, stream>>>(
        ctxb, owb, out_b, pvf, latent_out);
}
